// Model_24730421690432
// MI455X (gfx1250) — compile-verified, matrix operand order checked by reading, not run on hardware
//
#include <hip/hip_runtime.h>
#include <hip/hip_bf16.h>

typedef __attribute__((ext_vector_type(16))) _Float16 v16h;
typedef __attribute__((ext_vector_type(2)))  __fp16   v2hp;   // matches cvt_pkrtz return
typedef __attribute__((ext_vector_type(8)))  float    v8f;

#define EPSV 1e-8f

// ---- workspace layout (element offsets) ----
// floats:
//   z_ws  : [0,            524288)   B*T*128
//   gi_ws : [524288,      2621440)   B*32*512  (bg1 folded in)
//   gj_ws : [2621440,     4718592)   B*32*512
//   allg  : [4718592,     4751360)   B*256
// halfs (after float region):
//   bpack : 16nt * 16kt * 32lane * 16half  = 131072 _Float16 (WMMA B-fragment layout)
#define OFF_Z    0
#define OFF_GI   524288
#define OFF_GJ   2621440
#define OFF_ALLG 4718592
#define OFF_W2H  4751360   // float-element offset where half region starts

static __device__ __forceinline__ v8f wmma16(v16h a, v16h b, v8f c) {
  return __builtin_amdgcn_wmma_f32_16x16x32_f16(
      /*neg_a=*/false, a, /*neg_b=*/false, b,
      /*c_mod=*/(short)0, c, /*reuse_a=*/false, /*reuse_b=*/false);
}

// K-permutation inside a 32-wide k-step for the CDNA5 16-bit A/B fragment layout:
// lane<16 owns K {0..7,16..23}; lane>=16 owns K {8..15,24..31}; slot v holds a pair.
static __device__ __forceinline__ int kperm(int v, int hi) {
  return (hi ? 8 : 0) + (v < 4 ? 2 * v : 16 + 2 * (v - 4));
}

// ---------------- K0: pack Wg2 f32 -> f16 B-fragments ----------------
// bpack half index u = ((nt*16 + kt)*32 + lane)*16 + h
__global__ void k0_pack_b(const float* __restrict__ wg2, _Float16* __restrict__ bpack) {
  const int u = blockIdx.x * 256 + threadIdx.x;   // 0 .. 131071
  const int h    = u & 15;
  const int lane = (u >> 4) & 31;
  const int kt   = (u >> 9) & 15;
  const int nt   = u >> 13;
  const int n = nt * 16 + (lane & 15);
  const int k = kt * 32 + kperm(h >> 1, lane >> 4) + (h & 1);
  bpack[u] = (_Float16)wg2[k * 256 + n];
}

// ---------------- K1: encoder  h=relu(x@We1+be1), z=h@We2+be2 ----------------
__global__ void k1_encoder(const float* __restrict__ x_seq,
                           const float* __restrict__ We1, const float* __restrict__ be1,
                           const float* __restrict__ We2, const float* __restrict__ be2,
                           float* __restrict__ z_ws) {
  __shared__ float s_x[1024];
  __shared__ float s_h[256];
  const int row = blockIdx.x;           // b*32 + t
  const int tid = threadIdx.x;

  const float* xr = x_seq + (size_t)row * 1024;
  #pragma unroll
  for (int q = 0; q < 4; ++q) s_x[tid + q * 256] = xr[tid + q * 256];
  __syncthreads();

  float acc = be1[tid];
  #pragma unroll 4
  for (int d = 0; d < 1024; ++d) acc = fmaf(s_x[d], We1[d * 256 + tid], acc);
  s_h[tid] = fmaxf(acc, 0.0f);
  __syncthreads();

  if (tid < 128) {
    float zz = be2[tid];
    #pragma unroll 4
    for (int k = 0; k < 256; ++k) zz = fmaf(s_h[k], We2[k * 128 + tid], zz);
    z_ws[(size_t)row * 128 + tid] = zz;
  }
}

// ---------------- K2: temporal norm + tag + gi/gj ----------------
__global__ void k2_norm_gigj(const float* __restrict__ z_ws,
                             const float* __restrict__ gamma, const float* __restrict__ beta,
                             const float* __restrict__ Wg1, const float* __restrict__ bg1,
                             float* __restrict__ gi_ws, float* __restrict__ gj_ws) {
  __shared__ float s_zc[32][132];       // 129 used, padded
  const int b = blockIdx.x;
  const int tid = threadIdx.x;

  for (int idx = tid; idx < 4096; idx += 256) {
    s_zc[idx >> 7][idx & 127] = z_ws[(size_t)b * 4096 + idx];
  }
  if (tid < 32) s_zc[tid][128] = (float)tid;   // time tag
  __syncthreads();

  if (tid < 128) {
    const int c = tid;
    float mu = 0.f;
    #pragma unroll
    for (int t = 0; t < 32; ++t) mu += s_zc[t][c];
    mu *= (1.0f / 32.0f);
    float var = 0.f;
    #pragma unroll
    for (int t = 0; t < 32; ++t) { float d = s_zc[t][c] - mu; var = fmaf(d, d, var); }
    var *= (1.0f / 31.0f);               // ddof=1
    const float inv = gamma[c] * __frsqrt_rn(var + EPSV);
    const float sh  = beta[c] - mu * inv;
    #pragma unroll
    for (int t = 0; t < 32; ++t) s_zc[t][c] = fmaf(s_zc[t][c], inv, sh);
  }
  __syncthreads();

  // gi = zc @ Wg1[:129] + bg1 (folded), gj = zc @ Wg1[129:]
  for (int out = tid; out < 32 * 512; out += 256) {
    const int t = out >> 9, col = out & 511;
    float ai = bg1[col], aj = 0.f;
    #pragma unroll 4
    for (int c = 0; c < 129; ++c) {
      const float zv = s_zc[t][c];
      ai = fmaf(zv, Wg1[c * 512 + col], ai);
      aj = fmaf(zv, Wg1[(129 + c) * 512 + col], aj);
    }
    gi_ws[(size_t)b * 16384 + out] = ai;
    gj_ws[(size_t)b * 16384 + out] = aj;
  }
}

// ---------------- K3: pairwise relation core (WMMA) ----------------
// per block: one b.  sum_{i,j} relu( relu(gi[i]+gj[j]+bg1) @ Wg2 + bg2 )
// LDS: s_afrag 64KB (pre-packed A fragments for one 64-row chunk)
//      s_gi 64KB, s_gj 64KB  -> 192KB dynamic
__global__ void __launch_bounds__(256)
k3_pairwise(const float* __restrict__ gi_ws, const float* __restrict__ gj_ws,
            const _Float16* __restrict__ bpack, const float* __restrict__ bg2,
            float* __restrict__ allg) {
  extern __shared__ __align__(32) char smem_raw[];
  _Float16* s_afrag = (_Float16*)smem_raw;            // 4q*16kt*32lane*16h = 32768 halves
  float*    s_gi    = (float*)(smem_raw + 65536);     // 32*512
  float*    s_gj    = (float*)(smem_raw + 131072);    // 32*512

  const int b    = blockIdx.x;
  const int tid  = threadIdx.x;
  const int lane = tid & 31;
  const int wave = tid >> 5;            // 0..7
  const int lm   = lane & 15;

  // warm L2 for the packed f16 weights (global_prefetch_b8)
  __builtin_prefetch(bpack + tid * 512, 0, 0);

  for (int idx = tid; idx < 32 * 512; idx += 256) {
    s_gi[idx] = gi_ws[(size_t)b * 16384 + idx];   // bg1 already folded in
    s_gj[idx] = gj_ws[(size_t)b * 16384 + idx];
  }
  __syncthreads();

  const int nt0 = wave * 2, nt1 = wave * 2 + 1;   // this wave's two N tiles
  const float bgv0 = bg2[nt0 * 16 + lm], bgv1 = bg2[nt1 * 16 + lm];
  float cs0 = 0.f, cs1 = 0.f;                     // per-lane column partial sums

  // builder thread's fixed slot within fragments
  const int bv  = tid & 7;              // packed pair slot 0..7
  const int bln = (tid >> 3) & 31;      // fragment lane 0..31
  const int bhi = bln >> 4, blm = bln & 15;
  const int bkp = kperm(bv, bhi);       // k offset within a 32-wide k-step

  const v16h* ap = (const v16h*)s_afrag;
  const v16h* bp = (const v16h*)bpack;

  for (int rc = 0; rc < 16; ++rc) {               // 16 chunks x 4 row-tiles x 16 rows
    // ---- cooperative A-fragment build: each element computed once per block ----
    #pragma unroll
    for (int q = 0; q < 4; ++q) {
      const int r = rc * 64 + q * 16 + blm;       // pair row 0..1023
      const float* gr = s_gi + (r >> 5) * 512;    // i = r/32
      const float* hr = s_gj + (r & 31) * 512;    // j = r%32
      #pragma unroll 4
      for (int kt = 0; kt < 16; ++kt) {
        const int k0 = kt * 32 + bkp;
        const float f0 = fmaxf(gr[k0]     + hr[k0],     0.0f);
        const float f1 = fmaxf(gr[k0 + 1] + hr[k0 + 1], 0.0f);
        // v_cvt_pk_rtz_f16_f32; store slot u = tid + (q*16+kt)*256
        ((v2hp*)s_afrag)[((q * 16 + kt) * 32 + bln) * 8 + bv] =
            __builtin_amdgcn_cvt_pkrtz(f0, f1);
      }
    }
    __syncthreads();

    // ---- consume: pure wide loads + WMMA ----
    v8f acc[8];
    #pragma unroll
    for (int q = 0; q < 8; ++q) acc[q] = (v8f){};

    for (int kt = 0; kt < 16; ++kt) {
      const v16h bf0 = bp[(nt0 * 16 + kt) * 32 + lane];
      const v16h bf1 = bp[(nt1 * 16 + kt) * 32 + lane];
      v16h af[4];
      #pragma unroll
      for (int q = 0; q < 4; ++q) af[q] = ap[(q * 16 + kt) * 32 + lane];
      #pragma unroll
      for (int q = 0; q < 4; ++q) {
        acc[q]     = wmma16(af[q], bf0, acc[q]);
        acc[4 + q] = wmma16(af[q], bf1, acc[4 + q]);
      }
    }

    // relu(+bg2) and fold into column sums (each lane covers 8 of 16 rows)
    #pragma unroll
    for (int q = 0; q < 4; ++q) {
      #pragma unroll
      for (int m = 0; m < 8; ++m) {
        cs0 += fmaxf(acc[q][m]     + bgv0, 0.0f);
        cs1 += fmaxf(acc[4 + q][m] + bgv1, 0.0f);
      }
    }
    __syncthreads();   // before next chunk's build overwrites s_afrag
  }

  // combine lane l with lane l+16 (same output column, other 8 rows)
  cs0 += __shfl_down(cs0, 16, 32);
  cs1 += __shfl_down(cs1, 16, 32);
  if (lane < 16) {
    allg[(size_t)b * 256 + nt0 * 16 + lane] = cs0;
    allg[(size_t)b * 256 + nt1 * 16 + lane] = cs1;
  }
}

// ---------------- K4: head + argmax ----------------
__global__ void k4_head(const float* __restrict__ allg,
                        const float* __restrict__ Wf, const float* __restrict__ bf,
                        const float* __restrict__ Wy, const float* __restrict__ by,
                        float* __restrict__ out) {
  __shared__ float s_g[256];
  __shared__ float s_f[256];
  __shared__ float s_y[4];
  const int b = blockIdx.x, tid = threadIdx.x;

  s_g[tid] = allg[(size_t)b * 256 + tid];
  __syncthreads();

  float acc = bf[tid];
  #pragma unroll 4
  for (int g = 0; g < 256; ++g) acc = fmaf(s_g[g], Wf[g * 256 + tid], acc);
  s_f[tid] = fmaxf(acc, 0.0f);
  __syncthreads();

  if (tid < 4) {
    float y = by[tid];
    #pragma unroll 4
    for (int k = 0; k < 256; ++k) y = fmaf(s_f[k], Wy[k * 4 + tid], y);
    s_y[tid] = y;
    out[b * 4 + tid] = y;
  }
  __syncthreads();

  if (tid == 0) {
    int best = 0;
    float bv2 = s_y[0];
    #pragma unroll
    for (int c = 1; c < 4; ++c) if (s_y[c] > bv2) { bv2 = s_y[c]; best = c; }
    out[512 + b] = (float)best;          // y_pred, stored in output dtype (f32)
  }
}

extern "C" void kernel_launch(void* const* d_in, const int* in_sizes, int n_in,
                              void* d_out, int out_size, void* d_ws, size_t ws_size,
                              hipStream_t stream) {
  const float* x_seq = (const float*)d_in[0];
  const float* We1   = (const float*)d_in[1];
  const float* be1   = (const float*)d_in[2];
  const float* We2   = (const float*)d_in[3];
  const float* be2   = (const float*)d_in[4];
  const float* gamma = (const float*)d_in[5];
  const float* beta  = (const float*)d_in[6];
  const float* Wg1   = (const float*)d_in[7];
  const float* bg1   = (const float*)d_in[8];
  const float* Wg2   = (const float*)d_in[9];
  const float* bg2   = (const float*)d_in[10];
  const float* Wf    = (const float*)d_in[11];
  const float* bf    = (const float*)d_in[12];
  const float* Wy    = (const float*)d_in[13];
  const float* by    = (const float*)d_in[14];

  float* ws   = (float*)d_ws;
  float* z_ws = ws + OFF_Z;
  float* gi   = ws + OFF_GI;
  float* gj   = ws + OFF_GJ;
  float* allg = ws + OFF_ALLG;
  _Float16* bpack = (_Float16*)(ws + OFF_W2H);
  float* out = (float*)d_out;

  k0_pack_b<<<512, 256, 0, stream>>>(Wg2, bpack);
  k1_encoder<<<128 * 32, 256, 0, stream>>>(x_seq, We1, be1, We2, be2, z_ws);
  k2_norm_gigj<<<128, 256, 0, stream>>>(z_ws, gamma, beta, Wg1, bg1, gi, gj);
  k3_pairwise<<<128, 256, 196608, stream>>>(gi, gj, bpack, bg2, allg);
  k4_head<<<128, 256, 0, stream>>>(allg, Wf, bf, Wy, by, out);
}
